// LSTMModel_2302102471418
// MI455X (gfx1250) — matrix operand order repaired and checked by reading, not compiled, not run on hardware
//
#include <hip/hip_runtime.h>
#include <hip/hip_bf16.h>

// ---------------------------------------------------------------------------
// 2-layer LSTM (H=512, B=256, T=512, DIN=64) + FC head for MI455X (gfx1250).
// bf16 WMMA (16x16x32, fp32 accum) persistent recurrence; packed fragment-
// major weights stream from L2; h state in LDS, C state in registers.
// TDM tensor_load_to_lds stages layer-1 input tiles (LDS-padding descriptor),
// async LDS->global b128 stores export layer-0 h, WGP-scope weight prefetch.
// ---------------------------------------------------------------------------

typedef __attribute__((ext_vector_type(16))) __bf16 v16bf;
typedef __attribute__((ext_vector_type(8)))  __bf16 v8bf;
typedef __attribute__((ext_vector_type(4)))  __bf16 v4bf;
typedef __attribute__((ext_vector_type(8)))  float  v8f;
typedef __attribute__((ext_vector_type(4)))  unsigned int u32x4;
typedef __attribute__((ext_vector_type(4)))  int i32x4;
typedef __attribute__((ext_vector_type(8)))  int i32x8;

static constexpr int HD    = 512;   // hidden
static constexpr int BATCH = 256;
static constexpr int SEQ   = 512;
static constexpr int DINC  = 64;

#if __has_builtin(__builtin_amdgcn_tensor_load_to_lds)
#define HAS_TDM 1
#else
#define HAS_TDM 0
#endif
#if __has_builtin(__builtin_amdgcn_global_store_async_from_lds_b128)
#define HAS_ASYNC_ST 1
#else
#define HAS_ASYNC_ST 0
#endif

__device__ __forceinline__ void wait_tensorcnt0() {
#if __has_builtin(__builtin_amdgcn_s_wait_tensorcnt)
    __builtin_amdgcn_s_wait_tensorcnt(0);
#else
    asm volatile("s_wait_tensorcnt 0x0" ::: "memory");
#endif
}
__device__ __forceinline__ void wait_asynccnt0() {
#if __has_builtin(__builtin_amdgcn_s_wait_asynccnt)
    __builtin_amdgcn_s_wait_asynccnt(0);
#else
    asm volatile("s_wait_asynccnt 0x0" ::: "memory");
#endif
}

__device__ __forceinline__ float fsig(float x) {
    return __builtin_amdgcn_rcpf(1.0f + __expf(-x));
}
__device__ __forceinline__ float ftanh_fast(float x) {
    return 1.0f - 2.0f * __builtin_amdgcn_rcpf(1.0f + __expf(2.0f * x));
}

__device__ __forceinline__ v16bf ldb16(const __bf16* p) {
    const v8bf* q = (const v8bf*)p;
    return __builtin_shufflevector(q[0], q[1],
        0,1,2,3,4,5,6,7,8,9,10,11,12,13,14,15);
}
__device__ __forceinline__ v8f wmma_bf16(v16bf a, v16bf b, v8f c) {
    return __builtin_amdgcn_wmma_f32_16x16x32_bf16(
        false, a, false, b, (short)0, c, false, false);
}

// ---------------------------------------------------------------------------
// Pack combined [W_hh (4H x 512) | W_ih (4H x kin)] -> bf16, B-fragment-major:
//   halfIdx = ((g*32 + nt)*KC + kc)*512 + lane*16 + j*2 + h
//   n = g*512 + nt*16 + (lane&15); k = kc*32 + (lane>=16?16:0) + 2j + h
// ---------------------------------------------------------------------------
__global__ void pack_weights_kernel(const float* __restrict__ Whh,
                                    const float* __restrict__ Wih,
                                    int kin, __bf16* __restrict__ pack) {
    const int KC = (HD + kin) >> 5;
    long long idx = (long long)blockIdx.x * blockDim.x + threadIdx.x;
    const long long total = (long long)4 * HD * (HD + kin);
    if (idx >= total) return;
    int hb   = (int)(idx & 1);  long long r = idx >> 1;
    int j    = (int)(r & 7);    r >>= 3;
    int lane = (int)(r & 31);   r >>= 5;
    int kc   = (int)(r % KC);   r /= KC;
    int nt   = (int)(r & 31);   r >>= 5;
    int g    = (int)r;
    int n = g * HD + nt * 16 + (lane & 15);
    int k = kc * 32 + ((lane >= 16) ? 16 : 0) + j * 2 + hb;
    float v = (k < HD) ? Whh[(long long)n * HD + k]
                       : Wih[(long long)n * kin + (k - HD)];
    pack[idx] = (__bf16)v;
}

__global__ void bias_sum_kernel(const float* __restrict__ a,
                                const float* __restrict__ b,
                                float* __restrict__ out) {
    int i = blockIdx.x * blockDim.x + threadIdx.x;
    if (i < 4 * HD) out[i] = a[i] + b[i];
}

// ---------------------------------------------------------------------------
// Persistent LSTM layer. grid = BATCH/16 blocks of 256 threads (8 wave32).
// hlds [16][520] = h_{t-1} (bf16), xlds [16][KIN+8] = staged input x_t.
// Wave w computes column tiles nt = 4w..4w+3, all 4 gates each.
// ---------------------------------------------------------------------------
template <int KIN>
__global__ __launch_bounds__(256)
void lstm_layer_kernel(const __bf16* __restrict__ wpack,
                       const float*  __restrict__ bias,   // b_ih + b_hh [4H]
                       const float*  __restrict__ x0,     // KIN==64: [B,T,64]
                       const __bf16* __restrict__ hin,    // KIN==512: [T,B,512]
                       __bf16*       __restrict__ hout,   // layer0: [T,B,512]
                       __bf16*       __restrict__ hlast)  // layer1: [B,512]
{
    constexpr int KC   = (HD + KIN) / 32;    // combined K chunks
    constexpr int KCX  = KIN / 32;           // x-part chunks
    constexpr int RSH  = HD + 8;             // h row stride (halfs), 1040 B
    constexpr int RSX  = KIN + 8;            // x row stride (halfs)
    constexpr size_t GST = (size_t)32 * KC * 512;  // per-gate stride (halfs)

    __shared__ __bf16 hlds[16 * RSH];
    __shared__ __bf16 xlds[16 * RSX];

    const int tid  = threadIdx.x;
    const int lane = tid & 31;
    const int wave = tid >> 5;
    const int b0   = blockIdx.x << 4;
    const int nl   = lane & 15;
    const int hiHalf = (lane >= 16) ? 1 : 0;

    for (int i = tid; i < 16 * RSH; i += 256) hlds[i] = (__bf16)0.0f;

    float bsum[4][4];
    #pragma unroll
    for (int jt = 0; jt < 4; ++jt) {
        const int nt = wave * 4 + jt;
        #pragma unroll
        for (int g = 0; g < 4; ++g)
            bsum[jt][g] = bias[g * HD + nt * 16 + nl];
    }

    float cst[4][8];
    #pragma unroll
    for (int jt = 0; jt < 4; ++jt)
        #pragma unroll
        for (int r = 0; r < 8; ++r) cst[jt][r] = 0.0f;

    __syncthreads();

    const int arowH = nl * RSH;
    const int arowX = nl * RSX;
    const int asel  = hiHalf ? 16 : 0;

    for (int t = 0; t < SEQ; ++t) {
        // ---- stage x_t into xlds ----
        if (KIN == 64) {
            const int row = tid >> 4, c4 = (tid & 15) * 4;
            const float4 xv = *(const float4*)(x0 +
                (size_t)(b0 + row) * (SEQ * DINC) + (size_t)t * DINC + c4);
            v4bf xb;
            xb[0] = (__bf16)xv.x; xb[1] = (__bf16)xv.y;
            xb[2] = (__bf16)xv.z; xb[3] = (__bf16)xv.w;
            *(v4bf*)(xlds + row * RSX + c4) = xb;
        } else {
#if HAS_TDM
            // TDM: DMA the 16x512 bf16 tile of hin[t] into xlds, padding the
            // LDS destination 16B after every 1024B row to match RSX=520.
            if (wave == 0) {
                const unsigned ldsByte =
                    (unsigned)(size_t)(const void*)&xlds[0];
                const unsigned long long ga = (unsigned long long)(size_t)
                    (const void*)(hin + (size_t)t * BATCH * HD + (size_t)b0 * HD);
                u32x4 g0;
                g0[0] = 1u;                                   // count=1
                g0[1] = ldsByte;                              // lds_addr
                g0[2] = (unsigned)(ga & 0xffffffffu);         // gaddr[31:0]
                g0[3] = (unsigned)((ga >> 32) & 0x01ffffffu)  // gaddr[56:32]
                      | (2u << 30);                           // type=2 (image)
                i32x8 g1;
                g1[0] = (1 << 16)      // data_size = 2B
                      | (1 << 20)      // pad_enable
                      | (7 << 22)      // pad_interval = 256 dwords (1024 B)
                      | (3 << 25);     // pad_amount   = 4 dwords  (16 B)
                g1[1] = (HD & 0xffff) << 16;        // tensor_dim0 = 512
                g1[2] = (BATCH & 0xffff) << 16;     // tensor_dim1 = 256
                g1[3] = (HD & 0xffff) << 16;        // tile_dim0 = 512
                g1[4] = 16;                         // tile_dim1 = 16 rows
                g1[5] = HD;                         // tensor_dim0_stride = 512
                g1[6] = 0;
                g1[7] = 0;
                i32x4 gz; gz[0] = 0; gz[1] = 0; gz[2] = 0; gz[3] = 0;
#if defined(__clang_major__) && (__clang_major__ >= 23)
                i32x8 gz8;
                #pragma unroll
                for (int q = 0; q < 8; ++q) gz8[q] = 0;
                __builtin_amdgcn_tensor_load_to_lds(g0, g1, gz, gz, gz8, 0);
#else
                __builtin_amdgcn_tensor_load_to_lds(g0, g1, gz, gz, 0);
#endif
                wait_tensorcnt0();
            }
#else
            const int row = tid >> 4, c0 = (tid & 15) * 32;
            const uint4* src = (const uint4*)(hin +
                ((size_t)t * BATCH + b0 + row) * HD + c0);
            uint4* dst = (uint4*)(xlds + row * RSX + c0);
            #pragma unroll
            for (int q = 0; q < 4; ++q) dst[q] = src[q];
#endif
        }
        __syncthreads();

        // ---- gates = [h | x] @ [W_hh | W_ih]^T + bsum, then LSTM cell ----
        v8bf hnb[4];
        #pragma unroll
        for (int jt = 0; jt < 4; ++jt) {
            const int nt = wave * 4 + jt;
            v8f a0{}, a1{}, a2{}, a3{};
            const size_t wbase = (size_t)nt * KC * 512 + (size_t)lane * 16;
            // h part: K chunks 0..15 from hlds
            for (int kc = 0; kc < HD / 32; ++kc) {
                const v8bf* ap = (const v8bf*)(hlds + arowH + kc * 32 + asel);
                const v16bf av = __builtin_shufflevector(ap[0], ap[1],
                    0,1,2,3,4,5,6,7,8,9,10,11,12,13,14,15);
                const __bf16* wp = wpack + wbase + (size_t)kc * 512;
                __builtin_prefetch(wp + 512, 0, 3);   // next chunk, WGP scope
                a0 = wmma_bf16(av, ldb16(wp          ), a0);
                a1 = wmma_bf16(av, ldb16(wp +     GST), a1);
                a2 = wmma_bf16(av, ldb16(wp + 2 * GST), a2);
                a3 = wmma_bf16(av, ldb16(wp + 3 * GST), a3);
            }
            // x part: K chunks 16..KC-1 from xlds
            for (int kx = 0; kx < KCX; ++kx) {
                const v8bf* ap = (const v8bf*)(xlds + arowX + kx * 32 + asel);
                const v16bf av = __builtin_shufflevector(ap[0], ap[1],
                    0,1,2,3,4,5,6,7,8,9,10,11,12,13,14,15);
                const __bf16* wp = wpack + wbase + (size_t)(HD / 32 + kx) * 512;
                a0 = wmma_bf16(av, ldb16(wp          ), a0);
                a1 = wmma_bf16(av, ldb16(wp +     GST), a1);
                a2 = wmma_bf16(av, ldb16(wp + 2 * GST), a2);
                a3 = wmma_bf16(av, ldb16(wp + 3 * GST), a3);
            }
            #pragma unroll
            for (int r = 0; r < 8; ++r) {
                const float iv = a0[r] + bsum[jt][0];
                const float fv = a1[r] + bsum[jt][1];
                const float gv = a2[r] + bsum[jt][2];
                const float ov = a3[r] + bsum[jt][3];
                const float cn = fsig(fv) * cst[jt][r] + fsig(iv) * ftanh_fast(gv);
                cst[jt][r] = cn;
                hnb[jt][r] = (__bf16)(fsig(ov) * ftanh_fast(cn));
            }
        }

        __syncthreads();   // all A-fragment reads of h_{t-1} done

        // ---- write h_t back to LDS (D-tile layout -> row-major bf16) ----
        #pragma unroll
        for (int jt = 0; jt < 4; ++jt) {
            const int ncol = (wave * 4 + jt) * 16 + nl;
            #pragma unroll
            for (int r = 0; r < 8; ++r) {
                const int m = r + (hiHalf ? 8 : 0);
                hlds[m * RSH + ncol] = hnb[jt][r];
            }
        }

        // ---- export h_t (layer0: every t; layer1: last t only) ----
        if (KIN == 64) {
            __syncthreads();
            const int row = tid >> 4, c0 = (tid & 15) * 32;
            const __bf16* srcp = hlds + row * RSH + c0;
            __bf16* dstp = hout + ((size_t)t * BATCH + b0 + row) * HD + c0;
#if HAS_ASYNC_ST
            #pragma unroll
            for (int q = 0; q < 4; ++q) {
                __attribute__((address_space(1))) i32x4* gdst =
                    (__attribute__((address_space(1))) i32x4*)
                        (size_t)(const void*)(dstp + q * 8);
                __attribute__((address_space(3))) i32x4* lsrc =
                    (__attribute__((address_space(3))) i32x4*)
                        (size_t)(unsigned)(size_t)(const void*)(srcp + q * 8);
                __builtin_amdgcn_global_store_async_from_lds_b128(
                    gdst, lsrc, 0, 0);
            }
            wait_asynccnt0();
#else
            const uint4* src = (const uint4*)srcp;
            uint4* dst = (uint4*)dstp;
            #pragma unroll
            for (int q = 0; q < 4; ++q) dst[q] = src[q];
#endif
        } else if (t == SEQ - 1) {
            __syncthreads();
            const int row = tid >> 4, c0 = (tid & 15) * 32;
            const uint4* src = (const uint4*)(hlds + row * RSH + c0);
            uint4* dst = (uint4*)(hlast + (size_t)(b0 + row) * HD + c0);
            #pragma unroll
            for (int q = 0; q < 4; ++q) dst[q] = src[q];
        }
    }
}

// out[b] = dot(h2_last[b,:], fc_W) + fc_b   (DOUT == 1)
__global__ void fc_kernel(const __bf16* __restrict__ h,
                          const float* __restrict__ W,
                          const float* __restrict__ b,
                          float* __restrict__ out) {
    const int bi = blockIdx.x * blockDim.x + threadIdx.x;
    if (bi >= BATCH) return;
    float acc = 0.0f;
    const __bf16* hp = h + (size_t)bi * HD;
    #pragma unroll 8
    for (int k = 0; k < HD; ++k) acc += (float)hp[k] * W[k];
    out[bi] = acc + b[0];
}

extern "C" void kernel_launch(void* const* d_in, const int* in_sizes, int n_in,
                              void* d_out, int out_size, void* d_ws, size_t ws_size,
                              hipStream_t stream) {
    (void)in_sizes; (void)n_in; (void)out_size; (void)ws_size;

    const float* x     = (const float*)d_in[0];
    const float* W_ih0 = (const float*)d_in[1];
    const float* W_hh0 = (const float*)d_in[2];
    const float* b_ih0 = (const float*)d_in[3];
    const float* b_hh0 = (const float*)d_in[4];
    const float* W_ih1 = (const float*)d_in[5];
    const float* W_hh1 = (const float*)d_in[6];
    const float* b_ih1 = (const float*)d_in[7];
    const float* b_hh1 = (const float*)d_in[8];
    const float* fc_W  = (const float*)d_in[9];
    const float* fc_b  = (const float*)d_in[10];

    char* ws = (char*)d_ws;
    size_t off = 0;
    auto alloc = [&](size_t bytes) -> void* {
        off = (off + 255) & ~(size_t)255;
        void* p = ws + off;
        off += bytes;
        return p;
    };

    __bf16* w0p = (__bf16*)alloc((size_t)4 * HD * (HD + DINC) * 2); // 2.25 MB
    __bf16* w1p = (__bf16*)alloc((size_t)4 * HD * (HD + HD)   * 2); // 4 MB
    float*  bs0 = (float*) alloc((size_t)4 * HD * 4);
    float*  bs1 = (float*) alloc((size_t)4 * HD * 4);
    __bf16* h1b = (__bf16*)alloc((size_t)SEQ * BATCH * HD * 2);     // 134 MB
    __bf16* h2l = (__bf16*)alloc((size_t)BATCH * HD * 2);

    const long long tot0 = (long long)4 * HD * (HD + DINC);
    const long long tot1 = (long long)4 * HD * (HD + HD);
    pack_weights_kernel<<<(int)((tot0 + 255) / 256), 256, 0, stream>>>(
        W_hh0, W_ih0, DINC, w0p);
    pack_weights_kernel<<<(int)((tot1 + 255) / 256), 256, 0, stream>>>(
        W_hh1, W_ih1, HD, w1p);
    bias_sum_kernel<<<(4 * HD + 255) / 256, 256, 0, stream>>>(b_ih0, b_hh0, bs0);
    bias_sum_kernel<<<(4 * HD + 255) / 256, 256, 0, stream>>>(b_ih1, b_hh1, bs1);

    lstm_layer_kernel<DINC><<<BATCH / 16, 256, 0, stream>>>(
        w0p, bs0, x, nullptr, h1b, nullptr);
    lstm_layer_kernel<HD><<<BATCH / 16, 256, 0, stream>>>(
        w1p, bs1, nullptr, h1b, nullptr, h2l);

    fc_kernel<<<1, 256, 0, stream>>>(h2l, fc_W, fc_b, (float*)d_out);
}
